// LabelEmbeddingAttention_7404523618981
// MI455X (gfx1250) — compile-verified, matrix operand order checked by reading, not run on hardware
//
#include <hip/hip_runtime.h>
#include <hip/hip_bf16.h>

// Problem constants (from reference): B,S,D,C
#define BB 8
#define SS 2048
#define DD 1024
#define CC 4096

// CDNA5 async global->LDS staging (GLOBAL_LOAD_ASYNC_TO_LDS_B128 / ASYNCcnt).
// Implemented with inline asm to sidestep the builtin's CUDA-LangAS pointer
// sema. Set to 0 to fall back to global_load_b128 + ds_store_b128 staging.
#define USE_ASYNC_LDS 1

typedef __bf16 bf16;
typedef __attribute__((ext_vector_type(16))) __bf16 v16bf;
typedef __attribute__((ext_vector_type(8)))  __bf16 v8bf;
typedef __attribute__((ext_vector_type(4)))  __bf16 v4bf;
typedef __attribute__((ext_vector_type(8)))  float  v8f;

#if USE_ASYNC_LDS
__device__ __forceinline__ void async_cp16(const void* g, void* l) {
    // LDS flat addresses carry the LDS byte offset in addr[31:0] (aperture
    // rule), so truncating the generic pointer yields the ds address.
    unsigned int       lds_off = (unsigned int)(uintptr_t)l;
    unsigned long long gaddr   = (unsigned long long)(uintptr_t)g;
    asm volatile("global_load_async_to_lds_b128 %0, %1, off"
                 :
                 : "v"(lds_off), "v"(gaddr)
                 : "memory");
}
__device__ __forceinline__ void async_wait() {
    asm volatile("s_wait_asynccnt 0x0" ::: "memory");
}
#endif

// ---------------------------------------------------------------------------
// f32 -> bf16 cast (4 elements / thread)
// ---------------------------------------------------------------------------
__global__ void __launch_bounds__(256) cast_f32_to_bf16(const float* __restrict__ in,
                                                        bf16* __restrict__ out, int n) {
    int i = (blockIdx.x * 256 + threadIdx.x) * 4;
    if (i + 3 < n) {
        float4 v = *(const float4*)(in + i);
        v4bf o;
        o[0] = (bf16)v.x; o[1] = (bf16)v.y; o[2] = (bf16)v.z; o[3] = (bf16)v.w;
        *(v4bf*)(out + i) = o;
    }
}

// ---------------------------------------------------------------------------
// x [B,S,D] f32  ->  xT [B,D,S] bf16   (32x32 LDS tiles, coalesced both ways)
// ---------------------------------------------------------------------------
__global__ void __launch_bounds__(256) transpose_cast_x(const float* __restrict__ x,
                                                        bf16* __restrict__ xt) {
    __shared__ float t[32][33];
    const int b  = blockIdx.z;
    const int s0 = blockIdx.y * 32;
    const int d0 = blockIdx.x * 32;
    const int tx = threadIdx.x & 31;
    const int ty = threadIdx.x >> 5;          // 0..7
    const float* xb = x + ((size_t)b * SS + s0) * DD + d0;
#pragma unroll
    for (int j = 0; j < 4; ++j) {
        int r = ty + 8 * j;
        t[r][tx] = xb[(size_t)r * DD + tx];
    }
    __syncthreads();
    bf16* xtb = xt + ((size_t)b * DD + d0) * SS + s0;
#pragma unroll
    for (int j = 0; j < 4; ++j) {
        int r = ty + 8 * j;                   // local d index
        xtb[(size_t)r * SS + tx] = (bf16)t[tx][r];
    }
}

// ---------------------------------------------------------------------------
// Load a 16x32 bf16 operand fragment from an LDS tile laid out [idx][32].
// Lane l in [0,16): idx = base+l, K chunks {0..7, 16..23}
// Lane l+16       : idx = base+l, K chunks {8..15, 24..31}
// ---------------------------------------------------------------------------
__device__ __forceinline__ v16bf load_frag(const bf16* tile, int base) {
    int lid = threadIdx.x & 31;
    int l = lid & 15, h = lid >> 4;
    const bf16* p = tile + (base + l) * 32 + h * 8;
    union { v16bf v; v8bf half[2]; } u;
    u.half[0] = *(const v8bf*)(p);
    u.half[1] = *(const v8bf*)(p + 16);
    return u.v;
}

// 8-wave (2x4) 128x128 tile compute step from staged LDS tiles
__device__ __forceinline__ void wmma_step(const bf16* as, const bf16* bs,
                                          int wm, int wn, v8f acc[4][2]) {
    v16bf bfrag[2];
#pragma unroll
    for (int j = 0; j < 2; ++j) bfrag[j] = load_frag(bs, wn * 32 + j * 16);
#pragma unroll
    for (int i = 0; i < 4; ++i) {
        v16bf afrag = load_frag(as, wm * 64 + i * 16);
#pragma unroll
        for (int j = 0; j < 2; ++j)
            acc[i][j] = __builtin_amdgcn_wmma_f32_16x16x32_bf16(
                false, afrag, false, bfrag[j], (short)0, acc[i][j], false, false);
    }
}

// ---------------------------------------------------------------------------
// GEMM1: scores[b] = Xbf[b] (SxD) * Lbf^T (DxC)  -> logits f32 [B,S,C]
// Block: 256 threads (8 waves, 2x4), tile 128(M=s) x 128(N=c), BK=32
// Double-buffered LDS; async global->LDS staging.
// ---------------------------------------------------------------------------
__global__ void __launch_bounds__(256) gemm_scores(const bf16* __restrict__ X,
                                                   const bf16* __restrict__ L,
                                                   float* __restrict__ Sc) {
    __shared__ bf16 As[2][128 * 32];
    __shared__ bf16 Bs[2][128 * 32];
    const int b  = blockIdx.z;
    const int s0 = blockIdx.x * 128;
    const int c0 = blockIdx.y * 128;
    const bf16* Xb = X + (size_t)b * SS * DD + (size_t)s0 * DD;
    const bf16* Lb = L + (size_t)c0 * DD;
    const int tid  = threadIdx.x;
    const int wave = tid >> 5;
    const int wm   = wave >> 2;
    const int wn   = wave & 3;

    v8f acc[4][2];
#pragma unroll
    for (int i = 0; i < 4; ++i)
#pragma unroll
        for (int j = 0; j < 2; ++j) acc[i][j] = (v8f)(0.0f);

#if USE_ASYNC_LDS
    // prologue: stage k0 = 0 into buffer 0
#pragma unroll
    for (int c = tid; c < 512; c += 256) {
        int row = c >> 2, off = (c & 3) * 8;
        async_cp16(Xb + (size_t)row * DD + off, &As[0][row * 32 + off]);
        async_cp16(Lb + (size_t)row * DD + off, &Bs[0][row * 32 + off]);
    }
    int buf = 0;
    for (int k0 = 0; k0 < DD; k0 += 32) {
        async_wait();
        __syncthreads();           // staged data visible; previous buffer free
        if (k0 + 32 < DD) {
#pragma unroll
            for (int c = tid; c < 512; c += 256) {
                int row = c >> 2, off = (c & 3) * 8;
                async_cp16(Xb + (size_t)row * DD + k0 + 32 + off, &As[buf ^ 1][row * 32 + off]);
                async_cp16(Lb + (size_t)row * DD + k0 + 32 + off, &Bs[buf ^ 1][row * 32 + off]);
            }
        }
        wmma_step(As[buf], Bs[buf], wm, wn, acc);
        buf ^= 1;
    }
#else
    for (int k0 = 0; k0 < DD; k0 += 32) {
#pragma unroll
        for (int c = tid; c < 512; c += 256) {
            int row = c >> 2, off = (c & 3) * 8;
            *(v8bf*)(&As[0][row * 32 + off]) = *(const v8bf*)(Xb + (size_t)row * DD + k0 + off);
            *(v8bf*)(&Bs[0][row * 32 + off]) = *(const v8bf*)(Lb + (size_t)row * DD + k0 + off);
        }
        __syncthreads();
        wmma_step(As[0], Bs[0], wm, wn, acc);
        __syncthreads();
    }
#endif

    const int lid = tid & 31, l = lid & 15, h = lid >> 4;
    float* out = Sc + (size_t)b * SS * CC;
#pragma unroll
    for (int i = 0; i < 4; ++i)
#pragma unroll
        for (int j = 0; j < 2; ++j) {
            int n = c0 + wn * 32 + j * 16 + l;
#pragma unroll
            for (int r = 0; r < 8; ++r) {
                int m = s0 + wm * 64 + i * 16 + h * 8 + r;
                out[(size_t)m * CC + n] = acc[i][j][r];
            }
        }
}

// ---------------------------------------------------------------------------
// Row softmax in place over C=4096, one 256-thread block per (b,s) row
// ---------------------------------------------------------------------------
__global__ void __launch_bounds__(256) softmax_rows(float* __restrict__ A) {
    __shared__ float red[256];
    float* p = A + (size_t)blockIdx.x * CC;
    const int tid = threadIdx.x;

    float vals[16];
    float mx = -3.4e38f;
#pragma unroll
    for (int i = 0; i < 16; ++i) {
        vals[i] = p[tid + i * 256];
        mx = fmaxf(mx, vals[i]);
    }
    red[tid] = mx; __syncthreads();
    for (int s = 128; s > 0; s >>= 1) {
        if (tid < s) red[tid] = fmaxf(red[tid], red[tid + s]);
        __syncthreads();
    }
    mx = red[0]; __syncthreads();

    float sum = 0.0f;
#pragma unroll
    for (int i = 0; i < 16; ++i) {
        vals[i] = __expf(vals[i] - mx);
        sum += vals[i];
    }
    red[tid] = sum; __syncthreads();
    for (int s = 128; s > 0; s >>= 1) {
        if (tid < s) red[tid] += red[tid + s];
        __syncthreads();
    }
    float inv = 1.0f / red[0];
#pragma unroll
    for (int i = 0; i < 16; ++i) p[tid + i * 256] = vals[i] * inv;
}

// ---------------------------------------------------------------------------
// GEMM2: output[b] = A[b]^T (CxS) * X[b] (SxD) -> [B,C,D] f32
//   A operand: f32 softmax probs, converted to bf16 while staging transposed
//              (2 threads per c-row, 16 consecutive k each -> 2x ds_store_b128)
//   B operand: pre-transposed xT bf16 [B,D,S] -> contiguous b128 staging
// Block tile: 128(M=c) x 128(N=d), BK=32 over S
// ---------------------------------------------------------------------------
__global__ void __launch_bounds__(256) gemm_out(const float* __restrict__ A,
                                                const bf16* __restrict__ XT,
                                                float* __restrict__ Out) {
    __shared__ bf16 As[128 * 32];   // [m=c][k=s]
    __shared__ bf16 Bs[128 * 32];   // [n=d][k=s]
    const int b  = blockIdx.z;
    const int c0 = blockIdx.x * 128;
    const int d0 = blockIdx.y * 128;
    const float* Ab  = A  + (size_t)b * SS * CC;
    const bf16*  XTb = XT + (size_t)b * DD * SS;
    const int tid  = threadIdx.x;
    const int wave = tid >> 5;
    const int wm   = wave >> 2;
    const int wn   = wave & 3;

    v8f acc[4][2];
#pragma unroll
    for (int i = 0; i < 4; ++i)
#pragma unroll
        for (int j = 0; j < 2; ++j) acc[i][j] = (v8f)(0.0f);

    const int m  = tid >> 1;           // 0..127 : c-row within tile
    const int kb = (tid & 1) * 16;     // 0 or 16: k chunk

    for (int s0 = 0; s0 < SS; s0 += 32) {
        // B tile: Bs[n][k] = xT[(d0+n)*S + s0+k] -- contiguous 16B chunks
#pragma unroll
        for (int c = tid; c < 512; c += 256) {
            int row = c >> 2, off = (c & 3) * 8;
#if USE_ASYNC_LDS
            async_cp16(XTb + (size_t)(d0 + row) * SS + s0 + off, &Bs[row * 32 + off]);
#else
            *(v8bf*)(&Bs[row * 32 + off]) =
                *(const v8bf*)(XTb + (size_t)(d0 + row) * SS + s0 + off);
#endif
        }

        // A^T tile: As[m][kb+j] = A[(s0+kb+j)*C + c0+m], coalesced along m per j
        union { v8bf v[2]; bf16 e[16]; } u;
        const float* ap = Ab + (size_t)(s0 + kb) * CC + c0 + m;
#pragma unroll
        for (int j = 0; j < 16; ++j) u.e[j] = (bf16)ap[(size_t)j * CC];
        *(v8bf*)(&As[m * 32 + kb])     = u.v[0];
        *(v8bf*)(&As[m * 32 + kb + 8]) = u.v[1];

#if USE_ASYNC_LDS
        async_wait();
#endif
        __syncthreads();
        wmma_step(As, Bs, wm, wn, acc);
        __syncthreads();
    }

    const int lid = tid & 31, l = lid & 15, h = lid >> 4;
    float* out = Out + (size_t)b * CC * DD;
#pragma unroll
    for (int i = 0; i < 4; ++i)
#pragma unroll
        for (int j = 0; j < 2; ++j) {
            int n = d0 + wn * 32 + j * 16 + l;
#pragma unroll
            for (int r = 0; r < 8; ++r) {
                int mm = c0 + wm * 64 + i * 16 + h * 8 + r;
                out[(size_t)mm * DD + n] = acc[i][j][r];
            }
        }
}

// ---------------------------------------------------------------------------
// Launch
// ---------------------------------------------------------------------------
extern "C" void kernel_launch(void* const* d_in, const int* in_sizes, int n_in,
                              void* d_out, int out_size, void* d_ws, size_t ws_size,
                              hipStream_t stream) {
    const float* x     = (const float*)d_in[0];   // [B,S,D]
    const float* label = (const float*)d_in[1];   // [C,D]

    float* out_bcd = (float*)d_out;                       // [B,C,D]
    float* A_bsc   = out_bcd + (size_t)BB * CC * DD;      // [B,S,C]

    bf16* x_bf  = (bf16*)d_ws;                            // [B,S,D] bf16
    bf16* l_bf  = x_bf + (size_t)BB * SS * DD;            // [C,D]   bf16
    bf16* xT_bf = l_bf + (size_t)CC * DD;                 // [B,D,S] bf16

    const int nx = BB * SS * DD;
    const int nl = CC * DD;
    cast_f32_to_bf16<<<(nx / 4 + 255) / 256, 256, 0, stream>>>(x, x_bf, nx);
    cast_f32_to_bf16<<<(nl / 4 + 255) / 256, 256, 0, stream>>>(label, l_bf, nl);
    transpose_cast_x<<<dim3(DD / 32, SS / 32, BB), 256, 0, stream>>>(x, xT_bf);

    gemm_scores<<<dim3(SS / 128, CC / 128, BB), 256, 0, stream>>>(x_bf, l_bf, A_bsc);

    softmax_rows<<<BB * SS, 256, 0, stream>>>(A_bsc);

    gemm_out<<<dim3(CC / 128, DD / 128, BB), 256, 0, stream>>>(A_bsc, xT_bf, out_bcd);
}
